// CustomTransformer_58445914964311
// MI455X (gfx1250) — matrix operand order repaired and checked by reading, not compiled, not run on hardware
//
#include <hip/hip_runtime.h>
#include <cstddef>
#include <cstdint>

// ---------------------------------------------------------------------------
// CDNA5 (gfx1250) wave32 WMMA transformer forward pass.
// All matmuls lower to v_wmma_f32_16x16x32_bf16 (fp32 accumulate).
// ---------------------------------------------------------------------------

typedef __attribute__((ext_vector_type(16))) __bf16 v16bf;
typedef __attribute__((ext_vector_type(4)))  __bf16 v4bf;
typedef __attribute__((ext_vector_type(8)))  float  v8f;

union FragU { v16bf v; uint4 q[2]; };

__device__ __forceinline__ unsigned short f2bf(float f) {
  return __builtin_bit_cast(unsigned short, (__bf16)f);   // native cvt, RNE
}

__device__ __forceinline__ void store4bf(unsigned short* p, float4 f) {
  v4bf b;
  b[0] = (__bf16)f.x; b[1] = (__bf16)f.y;
  b[2] = (__bf16)f.z; b[3] = (__bf16)f.w;
  *reinterpret_cast<uint2*>(p) = __builtin_bit_cast(uint2, b);
}

// A-matrix fragment (16x32 bf16): lane m=L%16, half=L/16.
// VGPR0-3: K = half*8 + 0..7 ; VGPR4-7: K = 16 + half*8 + 0..7   (ISA 7.12.2)
__device__ __forceinline__ FragU load_a_frag(const unsigned short* base, int stride,
                                             int m_base, int kk, int lane) {
  const int m = m_base + (lane & 15);
  const int half = lane >> 4;
  const unsigned short* p = base + m * stride + kk;
  FragU f;
  f.q[0] = *reinterpret_cast<const uint4*>(p + half * 8);
  f.q[1] = *reinterpret_cast<const uint4*>(p + 16 + half * 8);
  return f;
}

// B-matrix fragment (32x16 bf16) from LDS stored as [N][K] rows:
// lane n=L%16, half=L/16. VGPR0-3: K = half*8 + 0..7 ; VGPR4-7: K = 16 + half*8 + 0..7
__device__ __forceinline__ FragU load_b_frag(const unsigned short* base, int stride,
                                             int n_base, int kk, int lane) {
  const int n = n_base + (lane & 15);
  const int half = lane >> 4;
  const unsigned short* p = base + n * stride + kk;
  FragU f;
  f.q[0] = *reinterpret_cast<const uint4*>(p + half * 8);
  f.q[1] = *reinterpret_cast<const uint4*>(p + 16 + half * 8);
  return f;
}

__device__ __forceinline__ v8f wmma_bf16(const FragU& a, const FragU& b, v8f c) {
  return __builtin_amdgcn_wmma_f32_16x16x32_bf16(false, a.v, false, b.v,
                                                 (short)0, c, false, false);
}

__device__ __forceinline__ float wave_reduce(float s) {
  for (int o = 16; o; o >>= 1) s += __shfl_down(s, o, 32);
  return s;
}

// ---------------------------------------------------------------------------
// Generic GEMM: out[M,N] = act(A[M,K] * W[N,K]^T + bias)
//   MODE 0: store, MODE 1: exact GELU then store,
//   MODE 2: out += comb[row*8+expert] * (acc+bias)  (MoE combine)
// block = 256 threads (8 waves); tile 64x64; double-buffered LDS with a
// 2-deep global prefetch pipeline: gload(k+64) is issued only after tile
// k+32 has been converted+stored, so each global_load_b128 batch has a full
// iteration (barrier + ds_loads + 2 WMMA) of latency cover before its wait.
// grid (ceil(N/64), M/64)
// ---------------------------------------------------------------------------
template <int MODE, int KD>
__global__ __launch_bounds__(256) void k_gemm(
    const float* __restrict__ A, const float* __restrict__ W,
    const float* __restrict__ bias, float* __restrict__ out,
    int Ndim, const float* __restrict__ comb, int expert)
{
  __shared__ unsigned short As[2][64 * 48];  // stride 48 ushorts (96B, 16B-mult)
  __shared__ unsigned short Bs[2][64 * 48];
  const int t = threadIdx.x;
  const int wave = t >> 5, lane = t & 31;
  const int m_base = (wave & 3) * 16;
  const int nh = wave >> 2;                  // 0..1 -> N cols [nh*32, nh*32+32)
  const int tile_n = blockIdx.x, tile_m = blockIdx.y;
  const int lrow = t >> 3;                   // 0..31
  const int lcol = (t & 7) << 2;             // 0,4,...,28
  v8f acc0{0.f,0.f,0.f,0.f,0.f,0.f,0.f,0.f};
  v8f acc1{0.f,0.f,0.f,0.f,0.f,0.f,0.f,0.f};

  float4 ra[2], rw[2];
  auto gload = [&](int k0) {
#pragma unroll
    for (int p = 0; p < 2; ++p) {
      const int r = lrow + p * 32;
      ra[p] = *reinterpret_cast<const float4*>(
          A + (size_t)(tile_m * 64 + r) * KD + k0 + lcol);
      const int wn = tile_n * 64 + r;
      rw[p] = make_float4(0.f, 0.f, 0.f, 0.f);
      if (wn < Ndim)
        rw[p] = *reinterpret_cast<const float4*>(W + (size_t)wn * KD + k0 + lcol);
    }
  };
  auto lstore = [&](int buf) {
#pragma unroll
    for (int p = 0; p < 2; ++p) {
      const int r = lrow + p * 32;
      store4bf(&As[buf][r * 48 + lcol], ra[p]);
      store4bf(&Bs[buf][r * 48 + lcol], rw[p]);
    }
  };
  auto step = [&](int k0, int cur) {
    __syncthreads();                          // LDS[cur] visible; old reads done
    const FragU a  = load_a_frag(As[cur], 48, m_base, 0, lane);
    const FragU b0 = load_b_frag(Bs[cur], 48, nh * 32, 0, lane);
    const FragU b1 = load_b_frag(Bs[cur], 48, nh * 32 + 16, 0, lane);
    acc0 = wmma_bf16(a, b0, acc0);
    acc1 = wmma_bf16(a, b1, acc1);
    if (k0 + 32 < KD) {
      lstore(cur ^ 1);                        // consumes gload(k0+32)
      if (k0 + 64 < KD) gload(k0 + 64);       // prefetch two steps ahead
    }
  };

  gload(0);
  lstore(0);
  gload(32);                                  // KD >= 64 always (768/3072)
  for (int k0 = 0; k0 < KD; k0 += 64) {       // KD multiple of 64
    step(k0, 0);
    step(k0 + 32, 1);
  }

  const int half = lane >> 4;
#pragma unroll
  for (int j = 0; j < 2; ++j) {
    const v8f acc = j ? acc1 : acc0;
    const int gn = tile_n * 64 + nh * 32 + j * 16 + (lane & 15);
    if (gn >= Ndim) continue;
    const float bv = bias ? bias[gn] : 0.f;
#pragma unroll
    for (int r = 0; r < 8; ++r) {
      const int gm = tile_m * 64 + m_base + r + half * 8;
      float v = acc[r] + bv;
      if constexpr (MODE == 1)
        v = 0.5f * v * (1.f + erff(v * 0.70710678118654752f));
      float* dst = out + (size_t)gm * Ndim + gn;
      if constexpr (MODE == 2) *dst += comb[gm * 8 + expert] * v;
      else                     *dst = v;
    }
  }
}

// ---------------------------------------------------------------------------
// Causal attention, one workgroup per (batch, head). T=64, HEAD=96.
// q/k/v/o are [512][768] with head h at cols [h*96, h*96+96).
// ---------------------------------------------------------------------------
__global__ __launch_bounds__(256) void k_attn(
    const float* __restrict__ qc, const float* __restrict__ kc,
    const float* __restrict__ vc, float* __restrict__ oc)
{
  __shared__ unsigned short Abuf[64 * 112];  // Q[64][96] then Vt[96][64] (stride 72)
  __shared__ unsigned short Bbuf[64 * 112];  // K[64][96] then P[64][64]
  __shared__ float Ss[64 * 72];
  const int bh = blockIdx.x;
  const int rbase = (bh >> 3) * 64;          // batch row base
  const int cb = (bh & 7) * 96;              // head col base
  const int t = threadIdx.x, wave = t >> 5, lane = t & 31;

  // Load Q,K as bf16 (64x96 = 1536 float4, 6 per thread)
#pragma unroll
  for (int i = 0; i < 6; ++i) {
    const int e4 = t * 6 + i;
    const int row = e4 / 24, c4 = (e4 % 24) * 4;
    const float4 fq = *reinterpret_cast<const float4*>(
        qc + (size_t)(rbase + row) * 768 + cb + c4);
    const float4 fk = *reinterpret_cast<const float4*>(
        kc + (size_t)(rbase + row) * 768 + cb + c4);
    store4bf(&Abuf[row * 112 + c4], fq);
    store4bf(&Bbuf[row * 112 + c4], fk);
  }
  __syncthreads();

  // S = (Q K^T) * 96^-0.5 ; skip fully-masked upper tiles
  const int ti = wave & 3;
#pragma unroll
  for (int jj = 0; jj < 2; ++jj) {
    const int tj = (wave >> 2) * 2 + jj;
    if (tj > ti) continue;
    v8f acc{0.f,0.f,0.f,0.f,0.f,0.f,0.f,0.f};
    for (int kk = 0; kk < 96; kk += 32) {
      const FragU a = load_a_frag(Abuf, 112, ti * 16, kk, lane);
      const FragU b = load_b_frag(Bbuf, 112, tj * 16, kk, lane);
      acc = wmma_bf16(a, b, acc);
    }
    const int n = tj * 16 + (lane & 15), half = lane >> 4;
#pragma unroll
    for (int r = 0; r < 8; ++r)
      Ss[(ti * 16 + r + half * 8) * 72 + n] = acc[r] * 0.10206207261596575f;
  }
  __syncthreads();

  // causal softmax per row -> P (bf16) into Bbuf (K no longer needed)
  if (t < 64) {
    const int r = t;
    float mx = -1e30f;
    for (int c = 0; c <= r; ++c) mx = fmaxf(mx, Ss[r * 72 + c]);
    float sum = 0.f;
    for (int c = 0; c <= r; ++c) { float e = __expf(Ss[r * 72 + c] - mx); Ss[r * 72 + c] = e; sum += e; }
    const float inv = 1.f / sum;
    for (int c = 0; c < 64; ++c)
      Bbuf[r * 112 + c] = f2bf((c <= r) ? Ss[r * 72 + c] * inv : 0.f);
  }
  __syncthreads();

  // V transposed into Abuf: Vt[d][s], stride 72 (Q no longer needed)
#pragma unroll
  for (int i = 0; i < 24; ++i) {
    const int e = t * 24 + i;                // 0..6143
    const int s = e / 96, d = e % 96;
    Abuf[d * 72 + s] = f2bf(vc[(size_t)(rbase + s) * 768 + cb + d]);
  }
  __syncthreads();

  // O = P V : 4 M-tiles x 6 N-tiles = 24 tiles, 3 per wave
  for (int tt = wave; tt < 24; tt += 8) {
    const int mi = tt & 3, dj = tt >> 2;
    v8f acc{0.f,0.f,0.f,0.f,0.f,0.f,0.f,0.f};
    for (int kk = 0; kk < 64; kk += 32) {
      const FragU a = load_a_frag(Bbuf, 112, mi * 16, kk, lane);
      const FragU b = load_b_frag(Abuf, 72, dj * 16, kk, lane);
      acc = wmma_bf16(a, b, acc);
    }
    const int n = dj * 16 + (lane & 15), half = lane >> 4;
#pragma unroll
    for (int r = 0; r < 8; ++r)
      oc[(size_t)(rbase + mi * 16 + r + half * 8) * 768 + cb + n] = acc[r];
  }
}

// ---------------------------------------------------------------------------
// x = LayerNorm(x + y) in place. One WG (256 thr) per 768-wide row.
// ---------------------------------------------------------------------------
__global__ __launch_bounds__(256) void k_ln(
    float* __restrict__ x, const float* __restrict__ y,
    const float* __restrict__ w, const float* __restrict__ b, int has_res)
{
  __shared__ float red[8];
  const int row = blockIdx.x, t = threadIdx.x, wave = t >> 5, lane = t & 31;
  float vv[3];
  float s = 0.f;
#pragma unroll
  for (int i = 0; i < 3; ++i) {
    const int c = t + i * 256;
    float xv = x[(size_t)row * 768 + c];
    if (has_res) xv += y[(size_t)row * 768 + c];
    vv[i] = xv; s += xv;
  }
  s = wave_reduce(s);
  if (lane == 0) red[wave] = s;
  __syncthreads();
  float tot = 0.f;
#pragma unroll
  for (int i = 0; i < 8; ++i) tot += red[i];
  const float mean = tot * (1.f / 768.f);
  __syncthreads();
  float s2 = 0.f;
#pragma unroll
  for (int i = 0; i < 3; ++i) { const float d = vv[i] - mean; s2 += d * d; }
  s2 = wave_reduce(s2);
  if (lane == 0) red[wave] = s2;
  __syncthreads();
  float var = 0.f;
#pragma unroll
  for (int i = 0; i < 8; ++i) var += red[i];
  const float rstd = rsqrtf(var * (1.f / 768.f) + 1e-5f);
#pragma unroll
  for (int i = 0; i < 3; ++i) {
    const int c = t + i * 256;
    x[(size_t)row * 768 + c] = (vv[i] - mean) * rstd * w[c] + b[c];
  }
}

// ---------------------------------------------------------------------------
// Top-2 gate: comb[t][8] combine weights. One WG per token; wave e -> expert e.
// ---------------------------------------------------------------------------
__global__ __launch_bounds__(256) void k_gate(
    const float* __restrict__ x, const float* __restrict__ gw,
    const float* __restrict__ gb, float* __restrict__ comb)
{
  __shared__ float logit[8];
  __shared__ float cw[8];
  const int tk = blockIdx.x, t = threadIdx.x, wave = t >> 5, lane = t & 31;
  float p = 0.f;
  for (int i = lane; i < 768; i += 32)
    p += x[(size_t)tk * 768 + i] * gw[(size_t)wave * 768 + i];
  p = wave_reduce(p);
  if (lane == 0) logit[wave] = p + gb[wave];
  __syncthreads();
  if (t == 0) {
    float mx = -1e30f;
    for (int e = 0; e < 8; ++e) mx = fmaxf(mx, logit[e]);
    float pr[8]; float sum = 0.f;
    for (int e = 0; e < 8; ++e) { pr[e] = __expf(logit[e] - mx); sum += pr[e]; }
    for (int e = 0; e < 8; ++e) pr[e] /= sum;
    int i1 = 0;
    for (int e = 1; e < 8; ++e) if (pr[e] > pr[i1]) i1 = e;
    int i2 = -1;
    for (int e = 0; e < 8; ++e) { if (e == i1) continue; if (i2 < 0 || pr[e] > pr[i2]) i2 = e; }
    const float norm = pr[i1] + pr[i2];
    for (int e = 0; e < 8; ++e) cw[e] = 0.f;
    cw[i1] = pr[i1] / norm;
    cw[i2] = pr[i2] / norm;
  }
  __syncthreads();
  if (t < 8) comb[(size_t)tk * 8 + t] = cw[t];
}

// ---------------------------------------------------------------------------
// x[t] = tok_emb[idx[t]] + pos_emb[t % 64]
// ---------------------------------------------------------------------------
__global__ __launch_bounds__(256) void k_embed(
    const int* __restrict__ idx, const float* __restrict__ te,
    const float* __restrict__ pe, float* __restrict__ x)
{
  const int tk = blockIdx.x, t = threadIdx.x;
  const int tok = idx[tk];
  const int pos = tk & 63;
#pragma unroll
  for (int i = 0; i < 3; ++i) {
    const int c = t + i * 256;
    x[(size_t)tk * 768 + c] = te[(size_t)tok * 768 + c] + pe[(size_t)pos * 768 + c];
  }
}

// ---------------------------------------------------------------------------
extern "C" void kernel_launch(void* const* d_in, const int* in_sizes, int n_in,
                              void* d_out, int out_size, void* d_ws, size_t ws_size,
                              hipStream_t stream) {
  const int*   idx     = (const int*)  d_in[0];
  const float* tok_emb = (const float*)d_in[1];
  const float* pos_emb = (const float*)d_in[2];
  const float* wq      = (const float*)d_in[3];
  const float* wk      = (const float*)d_in[4];
  const float* wv      = (const float*)d_in[5];
  const float* wproj   = (const float*)d_in[6];
  const float* bproj   = (const float*)d_in[7];
  const float* gate_w  = (const float*)d_in[8];
  const float* gate_b  = (const float*)d_in[9];
  const float* w1      = (const float*)d_in[10];
  const float* b1      = (const float*)d_in[11];
  const float* w2      = (const float*)d_in[12];
  const float* b2      = (const float*)d_in[13];
  const float* ln1_w   = (const float*)d_in[14];
  const float* ln1_b   = (const float*)d_in[15];
  const float* ln2_w   = (const float*)d_in[16];
  const float* ln2_b   = (const float*)d_in[17];
  const float* lnf_w   = (const float*)d_in[18];
  const float* lnf_b   = (const float*)d_in[19];
  const float* lm_w    = (const float*)d_in[20];
  const float* lm_b    = (const float*)d_in[21];
  float* out = (float*)d_out;

  float* ws = (float*)d_ws;
  const size_t MC = 512ull * 768ull;
  float* x   = ws;
  float* y   = x + MC;
  float* qb  = y + MC;
  float* kb  = qb + MC;
  float* vb  = kb + MC;
  float* ob  = vb + MC;
  float* hb  = ob + MC;                    // 512*3072
  float* comb = hb + 512ull * 3072ull;     // 512*8

  const dim3 blk(256);
  const dim3 g768(12, 8);                  // N=768 tiles x M=512 tiles
  const dim3 gdff(48, 8);                  // N=3072
  const dim3 glm(2, 8);                    // N=99

  k_embed<<<512, blk, 0, stream>>>(idx, tok_emb, pos_emb, x);

  for (int l = 0; l < 12; ++l) {
    const size_t wstride = 589824ull;      // 768*768 per layer
    k_gemm<0, 768><<<g768, blk, 0, stream>>>(x, wq + l * wstride, nullptr, qb,
                                             768, nullptr, 0);
    k_gemm<0, 768><<<g768, blk, 0, stream>>>(x, wk + l * wstride, nullptr, kb,
                                             768, nullptr, 0);
    k_gemm<0, 768><<<g768, blk, 0, stream>>>(x, wv + l * wstride, nullptr, vb,
                                             768, nullptr, 0);
    k_attn<<<64, blk, 0, stream>>>(qb, kb, vb, ob);
    k_gemm<0, 768><<<g768, blk, 0, stream>>>(ob, wproj + l * wstride,
                                             bproj + l * 768, y, 768, nullptr, 0);
    k_ln<<<512, blk, 0, stream>>>(x, y, ln1_w + l * 768, ln1_b + l * 768, 1);
    k_gate<<<512, blk, 0, stream>>>(x, gate_w + l * 6144, gate_b + l * 8, comb);
    hipMemsetAsync(y, 0, MC * sizeof(float), stream);
    for (int e = 0; e < 8; ++e) {
      const size_t we = (size_t)(l * 8 + e);
      k_gemm<1, 768><<<gdff, blk, 0, stream>>>(x, w1 + we * 2359296ull,
                                               b1 + we * 3072, hb, 3072,
                                               nullptr, 0);          // GELU
      k_gemm<2, 3072><<<g768, blk, 0, stream>>>(hb, w2 + we * 2359296ull,
                                                b2 + we * 768, y, 768,
                                                comb, e);            // combine
    }
    k_ln<<<512, blk, 0, stream>>>(x, y, ln2_w + l * 768, ln2_b + l * 768, 1);
  }

  k_ln<<<512, blk, 0, stream>>>(x, nullptr, lnf_w, lnf_b, 0);
  k_gemm<0, 768><<<glm, blk, 0, stream>>>(x, lm_w, lm_b, out, 99, nullptr, 0);
}
